// ImprovedMPS_55482387530039
// MI455X (gfx1250) — compile-verified, matrix operand order checked by reading, not run on hardware
//
#include <hip/hip_runtime.h>

typedef __attribute__((ext_vector_type(16))) _Float16 v16h;
typedef __attribute__((ext_vector_type(8)))  float    v8f;

#define NUM_SITES 64
#define NUM_PAIRS (NUM_SITES / 2)
#define BOND      32
#define BATCH     65536
// fragment buffer in d_ws (f16): [pair:32][tile:8][lane:32][elem:16]  = 256 KB
// tile t covers fused-output rows n = 16t..16t+15, n = g*32 + d,
// g = t>>1 = p*2+q (which A_s,p * A_{s+1,q} product), d = 16*(t&1) + m.
#define FRAG_ELEMS (NUM_PAIRS * 8 * 32 * 16)

// ---------------------------------------------------------------------------
// Pre-pass: build fused pair matrices M_g = A[2s',:,p,:] @ A[2s'+1,:,q,:]
// in f32, emit as f16 WMMA A-operand fragments (transposed: row M = n, K = l).
// CDNA5 16-bit 16x32 A layout: lane L holds row M = L&15; halfword e maps to
// K = (e<8 ? e : e+8) + ((L>=16) ? 8 : 0).
// ---------------------------------------------------------------------------
__global__ void mps_pack_frags(const float* __restrict__ A,
                               _Float16* __restrict__ frag) {
    int idx = blockIdx.x * blockDim.x + threadIdx.x;
    if (idx >= FRAG_ELEMS) return;
    int sp   = idx >> 12;            // pair 0..31
    int rem  = idx & 4095;
    int t    = rem >> 9;             // tile 0..7
    int lane = (rem >> 4) & 31;
    int e    = rem & 15;
    int m    = lane & 15;
    int hi   = lane >> 4;
    int K    = (e < 8 ? e : e + 8) + (hi ? 8 : 0);   // left-bond index l
    int g    = t >> 1;               // pq group
    int p    = g >> 1;
    int q    = g & 1;
    int d    = 16 * (t & 1) + m;     // right-bond index of the pair
    int s0   = 2 * sp, s1 = 2 * sp + 1;

    const float* __restrict__ a0 = A + ((s0 * BOND + K) * 2 + p) * BOND; // over j
    float acc = 0.0f;
    #pragma unroll 8
    for (int j = 0; j < BOND; ++j)
        acc = fmaf(a0[j], A[((s1 * BOND + j) * 2 + q) * BOND + d], acc);
    frag[idx] = (_Float16)acc;
}

// ---------------------------------------------------------------------------
// Weighted combine of the 8 D-tiles for one batch group; result is already in
// next-step B-fragment layout (even tiles -> elems 0..7, odd -> 8..15).
// ---------------------------------------------------------------------------
__device__ __forceinline__ v16h combine8(v8f D0, v8f D1, v8f D2, v8f D3,
                                         v8f D4, v8f D5, v8f D6, v8f D7,
                                         float4 xv, float* zout) {
    const float w0 = xv.x * xv.z;   // g=0: p=0,q=0
    const float w1 = xv.x * xv.w;   // g=1: p=0,q=1
    const float w2 = xv.y * xv.z;   // g=2: p=1,q=0
    const float w3 = xv.y * xv.w;   // g=3: p=1,q=1
    v16h nl;
    #pragma unroll
    for (int r = 0; r < 8; ++r) {
        float z0 = w0 * D0[r];
        z0 = fmaf(w1, D2[r], z0);
        z0 = fmaf(w2, D4[r], z0);
        z0 = fmaf(w3, D6[r], z0);
        float z1 = w0 * D1[r];
        z1 = fmaf(w1, D3[r], z1);
        z1 = fmaf(w2, D5[r], z1);
        z1 = fmaf(w3, D7[r], z1);
        nl[r]     = (_Float16)z0;
        nl[r + 8] = (_Float16)z1;
        if (r == 0) *zout = z0;     // d==0 on lanes 0..15
    }
    return nl;
}

#define WMMA_F16(A_, B_) __builtin_amdgcn_wmma_f32_16x16x32_f16( \
        false, (A_), false, (B_), (short)0, cz, false, false)

// ---------------------------------------------------------------------------
// Main contraction. One wave owns TWO independent batch groups of 16 rows
// (32 batch rows total): the 8 A-fragments loaded per pair are reused for two
// B operands, halving L2 fragment traffic (~512 MB total) and doubling
// per-wave WMMA ILP (16 independent WMMAs between serial recurrence points).
// ---------------------------------------------------------------------------
__global__ void __launch_bounds__(256) mps_contract(
        const float* __restrict__ x,          // [B, S, 2] f32
        const _Float16* __restrict__ frag,    // packed fused-pair fragments
        float* __restrict__ out) {            // [B] f32
    const int lane  = threadIdx.x & 31;
    const int wave  = threadIdx.x >> 5;
    const int base  = blockIdx.x * 256 + wave * 32;
    const int bcol0 = base + (lane & 15);
    const int bcol1 = base + 16 + (lane & 15);

    const v16h* __restrict__ fptr  = (const v16h*)frag;    // (sp*8+t)*32+lane
    const float4* __restrict__ xp  = (const float4*)x;     // [B, 32] site-pairs

    // left0 = e0 (one-hot at l=0): B-fragment K=0 -> lanes 0..15, elem 0
    v16h leftA = {}, leftB = {};
    if (lane < 16) { leftA[0] = (_Float16)1.0f; leftB[0] = (_Float16)1.0f; }

    float zoutA = 0.0f, zoutB = 0.0f;

    for (int sp = 0; sp < NUM_PAIRS; ++sp) {
        // Prefetch next pair's fragment block (global_prefetch_b8)
        if (sp + 1 < NUM_PAIRS)
            __builtin_prefetch((const void*)(fptr + (sp + 1) * 8 * 32 + lane), 0, 1);

        const int fb = sp * 8 * 32 + lane;
        v16h a0 = fptr[fb + 0 * 32];
        v16h a1 = fptr[fb + 1 * 32];
        v16h a2 = fptr[fb + 2 * 32];
        v16h a3 = fptr[fb + 3 * 32];
        v16h a4 = fptr[fb + 4 * 32];
        v16h a5 = fptr[fb + 5 * 32];
        v16h a6 = fptr[fb + 6 * 32];
        v16h a7 = fptr[fb + 7 * 32];

        const float4 xvA = xp[bcol0 * NUM_PAIRS + sp];  // x[s0,0..1], x[s1,0..1]
        const float4 xvB = xp[bcol1 * NUM_PAIRS + sp];

        const v8f cz = {};
        // Group A
        v8f A0 = WMMA_F16(a0, leftA);
        v8f A1 = WMMA_F16(a1, leftA);
        v8f A2 = WMMA_F16(a2, leftA);
        v8f A3 = WMMA_F16(a3, leftA);
        v8f A4 = WMMA_F16(a4, leftA);
        v8f A5 = WMMA_F16(a5, leftA);
        v8f A6 = WMMA_F16(a6, leftA);
        v8f A7 = WMMA_F16(a7, leftA);
        // Group B (reuses the same A-fragments)
        v8f B0 = WMMA_F16(a0, leftB);
        v8f B1 = WMMA_F16(a1, leftB);
        v8f B2 = WMMA_F16(a2, leftB);
        v8f B3 = WMMA_F16(a3, leftB);
        v8f B4 = WMMA_F16(a4, leftB);
        v8f B5 = WMMA_F16(a5, leftB);
        v8f B6 = WMMA_F16(a6, leftB);
        v8f B7 = WMMA_F16(a7, leftB);

        leftA = combine8(A0, A1, A2, A3, A4, A5, A6, A7, xvA, &zoutA);
        leftB = combine8(B0, B1, B2, B3, B4, B5, B6, B7, xvB, &zoutB);
    }

    // Final left has right bond 1: only d=0 is meaningful (tile 0, r=0,
    // lanes 0..15), kept in f32.
    if (lane < 16) {
        out[bcol0] = zoutA;
        out[bcol1] = zoutB;
    }
}

extern "C" void kernel_launch(void* const* d_in, const int* in_sizes, int n_in,
                              void* d_out, int out_size, void* d_ws, size_t ws_size,
                              hipStream_t stream) {
    const float* x = (const float*)d_in[0];   // [65536, 64, 2]
    const float* A = (const float*)d_in[1];   // [64, 32, 2, 32]
    float* out = (float*)d_out;               // [65536]
    _Float16* frag = (_Float16*)d_ws;         // 256 KB of fused-pair fragments

    mps_pack_frags<<<FRAG_ELEMS / 256, 256, 0, stream>>>(A, frag);
    mps_contract<<<BATCH / 256, 256, 0, stream>>>(x, frag, out);
}
